// Decoder_33234456936648
// MI455X (gfx1250) — matrix operand order repaired and checked by reading, not compiled, not run on hardware
//
#include <hip/hip_runtime.h>
#include <math.h>

// ---------------------------------------------------------------------------
// 2-layer LSTM greedy decoder for MI455X (gfx1250, wave32, WMMA + TDM).
// One workgroup = 16 batch rows, all 40 steps in-kernel.
// - bf16 WMMA 16x16x32, f32 accumulation, 4 gate accumulators per hidden tile
//   interleaved so dependent WMMAs are 4 apart (hides the 16-bit WMMA hazard).
// - Weights pre-packed fragment-major: per-lane 32B contiguous, gate offsets
//   are compile-time constants folded into global_load_b128 offsets.
// - Embedding lookup = TDM gather (tensor_load_to_lds, 16x 16-bit row indices)
//   with LDS padding 16B per 1024B to spread DS banks.
// ---------------------------------------------------------------------------

typedef __attribute__((ext_vector_type(16))) __bf16 v16bf;
typedef __attribute__((ext_vector_type(8)))  float  v8f;
typedef unsigned int u32x4 __attribute__((ext_vector_type(4)));
typedef int          i32x4 __attribute__((ext_vector_type(4)));
typedef int          i32x8 __attribute__((ext_vector_type(8)));

#define LAYERS 2
#define HID    1024
#define BATCH  512
#define VOCAB  18
#define STEPS  40
#define GATES  (4 * HID)
#define SROW   1040               // padded LDS row (bf16): 2080 B, 16B aligned
#define KB_N   (HID / 32)         // 32 k-blocks
#define HT_N   (HID / 16)         // 64 hidden tiles
#define WAVES  8
#define HT_PW  (HT_N / WAVES)     // 8 tiles per wave
#define GSTRIDE 1048576           // elems between gate blocks in packed weights

__device__ __forceinline__ unsigned short f2bf(float f) {
    unsigned u = __builtin_bit_cast(unsigned, f);
    u = (u + 0x7FFFu + ((u >> 16) & 1u)) >> 16;
    return (unsigned short)u;
}
__device__ __forceinline__ float bf2f(unsigned short s) {
    unsigned u = ((unsigned)s) << 16;
    return __builtin_bit_cast(float, u);
}
__device__ __forceinline__ float sigm(float x) { return 1.0f / (1.0f + __expf(-x)); }
__device__ __forceinline__ float tanh_fast(float x) {
    x = fminf(fmaxf(x, -15.0f), 15.0f);
    float a = __expf(2.0f * x);
    return (a - 1.0f) / (a + 1.0f);
}
// padded-row element offset: +8 elems (16B) after every 512 elems (1024B)
__device__ __forceinline__ int koff(int k) { return k + ((k >= 512) ? 8 : 0); }

// ---------------------------------------------------------------------------
// Prep 1: fp32 [L,4H,H] -> bf16 fragment-major packed weights.
// ---------------------------------------------------------------------------
__global__ void pack_weights(const float* __restrict__ wih,
                             const float* __restrict__ whh,
                             unsigned short* __restrict__ pih,
                             unsigned short* __restrict__ phh) {
    const size_t N = (size_t)LAYERS * GATES * HID;
    for (size_t e = (size_t)blockIdx.x * blockDim.x + threadIdx.x; e < N;
         e += (size_t)gridDim.x * blockDim.x) {
        int j    = (int)(e & 15);
        int lane = (int)((e >> 4) & 31);
        int kb   = (int)((e >> 9) & 31);
        int ntg  = (int)((e >> 14) & 255);
        int l    = (int)(e >> 22);
        int n    = ntg * 16 + (lane & 15);
        int k    = kb * 32 + (lane >> 4) * 8 + (j & 7) + (j >> 3) * 16;
        size_t s = (size_t)l * GATES * HID + (size_t)n * HID + k;
        pih[e] = f2bf(wih[s]);
        phh[e] = f2bf(whh[s]);
    }
}

__global__ void init_misc(const float* __restrict__ emb,
                          unsigned short* __restrict__ emb_bf,
                          float* __restrict__ c_state) {
    const int NE = VOCAB * HID;
    const int NC = LAYERS * BATCH * HID;
    for (int i = blockIdx.x * blockDim.x + threadIdx.x; i < NE;
         i += gridDim.x * blockDim.x)
        emb_bf[i] = f2bf(emb[i]);
    for (int i = blockIdx.x * blockDim.x + threadIdx.x; i < NC;
         i += gridDim.x * blockDim.x)
        c_state[i] = 0.0f;
}

// ---------------------------------------------------------------------------
// Main persistent decoder: grid 32 x 256 threads (8 waves).
// ---------------------------------------------------------------------------
__global__ void lstm_decode(const unsigned short* __restrict__ wih_p,
                            const unsigned short* __restrict__ whh_p,
                            const unsigned short* __restrict__ emb_bf,
                            const float* __restrict__ bih,
                            const float* __restrict__ bhh,
                            const float* __restrict__ wfc,
                            const float* __restrict__ bfc,
                            float* __restrict__ c_state,
                            float* __restrict__ out_tok,
                            float* __restrict__ out_logp,
                            float* __restrict__ out_hf,
                            float* __restrict__ out_cf) {
    extern __shared__ char smem[];
    unsigned short* sh_in = (unsigned short*)smem;                 // 16*SROW bf16
    unsigned short* sh_h  = (unsigned short*)(smem + 33280);       // [par][l][16][SROW]
    float*          sh_fc = (float*)(smem + 166400);               // [16][20]
    int*            sh_tk = (int*)(smem + 167680);                 // [16]

    const int wg   = blockIdx.x;
    const int tid  = threadIdx.x;
    const int wave = tid >> 5;
    const int lane = tid & 31;
    const int ln   = lane & 15;
    const int hf   = lane >> 4;

    if (tid < 16) sh_tk[tid] = 0;
    for (int i = tid; i < 4 * 16 * SROW; i += 256) sh_h[i] = 0;

#pragma unroll 1
    for (int t = 0; t < STEPS; ++t) {
        const int par  = t & 1;
        const int npar = par ^ 1;

        __syncthreads();   // sh_tk ready

        // ---- embedding gather into sh_in via Tensor Data Mover ----
#if __has_builtin(__builtin_amdgcn_tensor_load_to_lds)
        if (wave == 0) {
            int tk[16];
#pragma unroll
            for (int i = 0; i < 16; ++i) tk[i] = sh_tk[i];

            const unsigned      lds_addr = (unsigned)(size_t)(void*)sh_in;
            const unsigned long long ga  = (unsigned long long)(size_t)emb_bf;

            u32x4 g0;
            g0[0] = 1u | (1u << 31);                       // count=1, gather_mode, 16b idx
            g0[1] = lds_addr;
            g0[2] = (unsigned)ga;                          // global_addr[31:0]
            g0[3] = (unsigned)((ga >> 32) & 0x01FFFFFFu) | (2u << 30); // [56:32], type=2

            i32x8 g1;
            g1[0] = (1 << 16)        // data_size = 2B
                  | (1 << 20)        // pad_enable
                  | (7 << 22)        // pad_interval: 256 DWORDs = 1024B
                  | (3 << 25);       // pad_amount:   4 DWORDs  = 16B
            g1[1] = (int)(1024u << 16);   // tensor_dim0 = 1024 (low 16 @ bit48)
            g1[2] = (int)((unsigned)VOCAB << 16); // tensor_dim1 = 18
            g1[3] = (int)(1024u << 16);   // tile_dim0 = 1024
            g1[4] = 16;                   // tile_dim1 = 16 gather indices
            g1[5] = 1024;                 // tensor_dim0_stride
            g1[6] = 0;
            g1[7] = 0;

            i32x4 g2, g3;
#pragma unroll
            for (int i = 0; i < 4; ++i) {
                g2[i] = (tk[2 * i] & 0xFFFF) | (tk[2 * i + 1] << 16);
                g3[i] = (tk[8 + 2 * i] & 0xFFFF) | (tk[8 + 2 * i + 1] << 16);
            }
#if defined(__clang_major__) && __clang_major__ >= 23
            i32x8 g4 = {0, 0, 0, 0, 0, 0, 0, 0};
            __builtin_amdgcn_tensor_load_to_lds(g0, g1, g2, g3, g4, 0);
#else
            __builtin_amdgcn_tensor_load_to_lds(g0, g1, g2, g3, 0);
#endif
            __builtin_amdgcn_s_wait_tensorcnt(0);
        }
#else
        for (int i = tid; i < 16 * HID; i += 256) {
            int r = i >> 10, k = i & (HID - 1);
            sh_in[r * SROW + koff(k)] = emb_bf[sh_tk[r] * HID + k];
        }
#endif
        __syncthreads();

#pragma unroll 1
        for (int l = 0; l < LAYERS; ++l) {
            const unsigned short* Ain =
                (l == 0) ? sh_in : (sh_h + (npar * 2 + 0) * 16 * SROW);
            const unsigned short* Ah  = sh_h + (par * 2 + l) * 16 * SROW;
            unsigned short*       Hout = sh_h + (npar * 2 + l) * 16 * SROW;
            const unsigned short* WihL = wih_p + (size_t)l * GATES * HID;
            const unsigned short* WhhL = whh_p + (size_t)l * GATES * HID;

#pragma unroll 1
            for (int tt = 0; tt < HT_PW; ++tt) {
                const int ht = wave * HT_PW + tt;
                const int n  = ht * 16 + ln;
                const int n8 = koff(n);

                v8f acc[4];
#pragma unroll
                for (int g = 0; g < 4; ++g) {
                    float bv = bih[l * GATES + g * HID + n] +
                               bhh[l * GATES + g * HID + n];
                    acc[g] = (v8f){bv, bv, bv, bv, bv, bv, bv, bv};
                }

                // per-lane fragment pointers; gate blocks at +g*GSTRIDE elems
                const unsigned short* pIH =
                    WihL + ((size_t)ht * KB_N * 32 + lane) * 16;
                const unsigned short* pHH =
                    WhhL + ((size_t)ht * KB_N * 32 + lane) * 16;
                const int abase = ln * SROW + hf * 8;

#pragma unroll 2
                for (int kb = 0; kb < KB_N; ++kb) {
                    const int ka = abase + kb * 32 + ((kb >= 16) ? 8 : 0);
                    v16bf ain, ah;
                    ((uint4*)&ain)[0] = *(const uint4*)(Ain + ka);
                    ((uint4*)&ain)[1] = *(const uint4*)(Ain + ka + 16);
                    ((uint4*)&ah)[0]  = *(const uint4*)(Ah + ka);
                    ((uint4*)&ah)[1]  = *(const uint4*)(Ah + ka + 16);

                    v16bf bi[4], bh[4];
#pragma unroll
                    for (int g = 0; g < 4; ++g) {
                        ((uint4*)&bi[g])[0] = *(const uint4*)(pIH + g * GSTRIDE);
                        ((uint4*)&bi[g])[1] = *(const uint4*)(pIH + g * GSTRIDE + 8);
                        ((uint4*)&bh[g])[0] = *(const uint4*)(pHH + g * GSTRIDE);
                        ((uint4*)&bh[g])[1] = *(const uint4*)(pHH + g * GSTRIDE + 8);
                    }
                    // 4 independent chains with ain, then ah: dependent WMMAs
                    // on the same accumulator are 4 apart.
#pragma unroll
                    for (int g = 0; g < 4; ++g)
                        acc[g] = __builtin_amdgcn_wmma_f32_16x16x32_bf16(
                            false, ain, false, bi[g], (short)0, acc[g], false, false);
#pragma unroll
                    for (int g = 0; g < 4; ++g)
                        acc[g] = __builtin_amdgcn_wmma_f32_16x16x32_bf16(
                            false, ah, false, bh[g], (short)0, acc[g], false, false);

                    __builtin_prefetch(pIH + 4 * 512, 0, 1);
                    __builtin_prefetch(pHH + 4 * 512, 0, 1);
                    pIH += 512;   // next k-block: 1024 B
                    pHH += 512;
                }

                // ---- LSTM pointwise on the C/D fragment ----
                const int mrow = hf * 8;
                const size_t cbase =
                    ((size_t)l * BATCH + wg * 16 + mrow) * HID + n;
                float cfr[8];
#pragma unroll
                for (int r = 0; r < 8; ++r)
                    cfr[r] = c_state[cbase + (size_t)r * HID];
#pragma unroll
                for (int r = 0; r < 8; ++r) {
                    float ig = acc[0][r], fg = acc[1][r];
                    float gg = acc[2][r], og = acc[3][r];
                    float cn = sigm(fg) * cfr[r] + sigm(ig) * tanh_fast(gg);
                    float hn = sigm(og) * tanh_fast(cn);
                    c_state[cbase + (size_t)r * HID] = cn;
                    Hout[(mrow + r) * SROW + n8] = f2bf(hn);
                    if (t == STEPS - 1) {
                        size_t o = (size_t)l * BATCH * HID +
                                   (size_t)(wg * 16 + mrow + r) * HID + n;
                        out_hf[o] = hn;
                        out_cf[o] = cn;
                    }
                }
            }
            __syncthreads();
        }

        // ---- FC (V=18), vectorized 8-wide ----
        const unsigned short* Hf = sh_h + (npar * 2 + 1) * 16 * SROW;
        for (int p = tid; p < 16 * VOCAB; p += 256) {
            const int r = p / VOCAB, v = p % VOCAB;
            const float* wv = wfc + (size_t)v * HID;
            const unsigned short* hrow = Hf + r * SROW;
            float s = bfc[v];
#pragma unroll 2
            for (int k = 0; k < HID; k += 8) {
                union { uint4 q; unsigned short us[8]; } u;
                u.q = *(const uint4*)(hrow + koff(k));
                const float4* wq = (const float4*)(wv + k);
                float4 w0 = wq[0], w1 = wq[1];
                s += bf2f(u.us[0]) * w0.x + bf2f(u.us[1]) * w0.y +
                     bf2f(u.us[2]) * w0.z + bf2f(u.us[3]) * w0.w +
                     bf2f(u.us[4]) * w1.x + bf2f(u.us[5]) * w1.y +
                     bf2f(u.us[6]) * w1.z + bf2f(u.us[7]) * w1.w;
            }
            sh_fc[r * 20 + v] = s;
        }
        __syncthreads();

        // ---- log_softmax + masked argmax ----
        if (tid < 16) {
            const int r = tid;
            float mx = -INFINITY;
            for (int v = 0; v < VOCAB; ++v) mx = fmaxf(mx, sh_fc[r * 20 + v]);
            float se = 0.0f;
            for (int v = 0; v < VOCAB; ++v) se += __expf(sh_fc[r * 20 + v] - mx);
            const float lse = mx + __logf(se);

            const int node_end = ((t >> 1) % 10) / 2 + 3;
            int best = 0;
            float bv = -INFINITY;
            for (int v = 0; v < VOCAB; ++v) {
                bool ok = (t & 1) ? (v >= 7) : (v >= 1 && v < node_end);
                float val = sh_fc[r * 20 + v] - lse;
                if (ok && val > bv) { bv = val; best = v; }
            }
            sh_tk[r] = best;
            out_tok[(size_t)t * BATCH + wg * 16 + r] = (float)best;
            if (t == STEPS - 1)
                for (int v = 0; v < VOCAB; ++v)
                    out_logp[(size_t)(wg * 16 + r) * VOCAB + v] =
                        sh_fc[r * 20 + v] - lse;
        }
        __syncthreads();
    }
}

// ---------------------------------------------------------------------------
extern "C" void kernel_launch(void* const* d_in, const int* in_sizes, int n_in,
                              void* d_out, int out_size, void* d_ws, size_t ws_size,
                              hipStream_t stream) {
    (void)in_sizes; (void)n_in; (void)out_size; (void)ws_size;
    const float* emb = (const float*)d_in[1];
    const float* wih = (const float*)d_in[2];
    const float* whh = (const float*)d_in[3];
    const float* bih = (const float*)d_in[4];
    const float* bhh = (const float*)d_in[5];
    const float* wfc = (const float*)d_in[6];
    const float* bfc = (const float*)d_in[7];

    char* ws = (char*)d_ws;
    const size_t WSZ = (size_t)LAYERS * GATES * HID * sizeof(unsigned short);
    unsigned short* wih_p  = (unsigned short*)ws;
    unsigned short* whh_p  = (unsigned short*)(ws + WSZ);
    unsigned short* emb_bf = (unsigned short*)(ws + 2 * WSZ);
    float*          c_st   = (float*)(ws + 2 * WSZ + VOCAB * HID * sizeof(unsigned short));

    float* outp     = (float*)d_out;
    float* out_tok  = outp;
    float* out_logp = outp + (size_t)STEPS * BATCH;
    float* out_hf   = out_logp + (size_t)BATCH * VOCAB;
    float* out_cf   = out_hf + (size_t)LAYERS * BATCH * HID;

    pack_weights<<<4096, 256, 0, stream>>>(wih, whh, wih_p, whh_p);
    init_misc<<<2048, 256, 0, stream>>>(emb, emb_bf, c_st);

    const size_t dyn_lds = 167744;  // sh_in + 4x sh_h (SROW=1040) + fc + tok
    lstm_decode<<<dim3(BATCH / 16), dim3(256), dyn_lds, stream>>>(
        wih_p, whh_p, emb_bf, bih, bhh, wfc, bfc, c_st,
        out_tok, out_logp, out_hf, out_cf);
}
